// OptimizedMoEAttention_27668179321498
// MI455X (gfx1250) — compile-verified, matrix operand order checked by reading, not run on hardware
//
#include <hip/hip_runtime.h>
#include <hip/hip_bf16.h>

#define B_  4
#define S_  2048
#define D_  1024
#define H_  16
#define HD_ 64
#define E_  4

typedef __bf16 bf16_t;
typedef __attribute__((ext_vector_type(16))) __bf16 v16bf;
typedef __attribute__((ext_vector_type(8)))  float  v8f;
typedef __attribute__((ext_vector_type(4)))  float  f32x4;

// Load one 16x32 bf16 A/B fragment (wave32 CDNA5 layout).
// Caller passes p = &M[row_for_this_lane][k0 + (lane>>4)*8]; we fetch
// elements [0..7] and [16..23] relative to p (two 16B chunks).
static __device__ __forceinline__ v16bf load_frag16(const bf16_t* p) {
  union { v16bf v; f32x4 q[2]; } u;
  u.q[0] = *(const f32x4*)(p);
  u.q[1] = *(const f32x4*)(p + 16);
  return u.v;
}

static __device__ __forceinline__ v8f wmma_bf16(v16bf a, v16bf b, v8f c) {
  return __builtin_amdgcn_wmma_f32_16x16x32_bf16(false, a, false, b,
                                                 (short)0, c, false, false);
}

// ---------------- router: logits -> softmax -> top2 -> combine weights ------
__global__ __launch_bounds__(256) void moe_router(
    const float* __restrict__ x, const float* __restrict__ rw,
    const float* __restrict__ rb, float* __restrict__ wtok) {
  int wave = threadIdx.x >> 5;
  int lane = threadIdx.x & 31;
  int t = blockIdx.x * 8 + wave;            // token index over B*S
  const float* xr = x + (size_t)t * D_;
  float acc[E_] = {0.f, 0.f, 0.f, 0.f};
  for (int d = lane; d < D_; d += 32) {
    float xv = xr[d];
#pragma unroll
    for (int e = 0; e < E_; ++e) acc[e] += xv * rw[d * E_ + e];
  }
#pragma unroll
  for (int mk = 16; mk >= 1; mk >>= 1)
#pragma unroll
    for (int e = 0; e < E_; ++e) acc[e] += __shfl_xor(acc[e], mk, 32);
  if (lane == 0) {
    float lg[E_], p[E_];
    float mx = -3.0e38f;
#pragma unroll
    for (int e = 0; e < E_; ++e) { lg[e] = acc[e] + rb[e]; mx = fmaxf(mx, lg[e]); }
    float s = 0.f;
#pragma unroll
    for (int e = 0; e < E_; ++e) { p[e] = __expf(lg[e] - mx); s += p[e]; }
#pragma unroll
    for (int e = 0; e < E_; ++e) p[e] /= s;
    int t1 = 0;
#pragma unroll
    for (int e = 1; e < E_; ++e) if (p[e] > p[t1]) t1 = e;
    int t2 = -1;
#pragma unroll
    for (int e = 0; e < E_; ++e)
      if (e != t1 && (t2 < 0 || p[e] > p[t2])) t2 = e;
    int b = t / S_, sx = t % S_;
#pragma unroll
    for (int e = 0; e < E_; ++e)
      wtok[((size_t)b * E_ + e) * S_ + sx] = (e == t1 || e == t2) ? p[e] : 0.f;
  }
}

// ---------------- per-(b,e) compaction via wave ballot scan -----------------
__global__ __launch_bounds__(32) void moe_compact(
    const float* __restrict__ wtok, int e, int* __restrict__ cnt,
    int* __restrict__ idx, float* __restrict__ gw) {
  int b = blockIdx.x, lane = threadIdx.x;
  const float* wr = wtok + ((size_t)b * E_ + e) * S_;
  int base = 0;
  for (int s0 = 0; s0 < S_; s0 += 32) {
    float w = wr[s0 + lane];
    bool m = w > 0.f;
    unsigned mask = (unsigned)__ballot(m);
    int pre = __popc(mask & ((1u << lane) - 1u));
    if (m) {
      idx[b * S_ + base + pre] = s0 + lane;
      gw[b * S_ + base + pre]  = w;
    }
    base += __popc(mask);
  }
  if (lane == 0) cnt[b] = base;
}

// ---------------- gather member tokens to bf16, zero-pad the rest -----------
__global__ __launch_bounds__(128) void moe_gather(
    const float* __restrict__ x, const int* __restrict__ cnt,
    const int* __restrict__ idx, bf16_t* __restrict__ xg) {
  int slot = blockIdx.x, b = blockIdx.y;
  int j0 = threadIdx.x * 8;                          // 128 threads * 8 = 1024
  bf16_t* dst = xg + ((size_t)b * S_ + slot) * D_ + j0;
  union { bf16_t h[8]; f32x4 q; } u;
  if (slot < cnt[b]) {
    const float* src = x + ((size_t)b * S_ + idx[b * S_ + slot]) * D_ + j0;
#pragma unroll
    for (int j = 0; j < 8; ++j) u.h[j] = (bf16_t)src[j];
  } else {
#pragma unroll
    for (int j = 0; j < 8; ++j) u.h[j] = (bf16_t)0.f;
  }
  *(f32x4*)dst = u.q;
}

// ---------------- weights: f32 [E][K][N] -> bf16 n-major [E][N][K] ----------
__global__ __launch_bounds__(128) void moe_wconvert(
    const float* __restrict__ w, bf16_t* __restrict__ wt) {
  int n = blockIdx.x, e = blockIdx.y;
  int k0 = threadIdx.x * 8;
  const float* we = w + (size_t)e * D_ * D_ + n;
  union { bf16_t h[8]; f32x4 q; } u;
#pragma unroll
  for (int j = 0; j < 8; ++j) u.h[j] = (bf16_t)we[(size_t)(k0 + j) * D_];
  *(f32x4*)(wt + ((size_t)e * D_ + n) * D_ + k0) = u.q;
}

// ---------------- 128x64-tile bf16 WMMA GEMM, 3 store modes -----------------
// Block = 4 waves; each wave owns a 32(M)x64(N) strip (8 accumulators).
// Manual ping-pong double buffering (x/y fragment sets) unrolled by 2 so the
// register allocator needs no rotation moves between iterations.
// mode 0: outB[b][m][n] = acc+bias (row-major bf16)         (Q,K)
// mode 1: outB[b][n][m] = acc+bias (transposed bf16)        (V -> V^T)
// mode 2: atomicAdd(out[b][idx[m]][n], gw[m]*(acc+bias))    (Wo + scatter)
__global__ __launch_bounds__(128) void moe_gemm(
    const bf16_t* __restrict__ Ag, const bf16_t* __restrict__ WT,
    const float* __restrict__ bias, const int* __restrict__ cnt,
    bf16_t* __restrict__ outB, float* __restrict__ outScat,
    const int* __restrict__ idx, const float* __restrict__ gw, int mode) {
  int b = blockIdx.z;
  int c = cnt[b];
  int m0 = blockIdx.x * 128;
  if (m0 >= c) return;
  int n0 = blockIdx.y * 64;
  int wave = threadIdx.x >> 5, lane = threadIdx.x & 31;
  int r = lane & 15, g = lane >> 4;
  const bf16_t* A0 = Ag + ((size_t)b * S_ + m0 + wave * 32 + r) * D_ + g * 8;
  const bf16_t* A1 = A0 + (size_t)16 * D_;
  const bf16_t* W0 = WT + ((size_t)(n0 + r)) * D_ + g * 8;
  const bf16_t* W1 = W0 + (size_t)16 * D_;
  const bf16_t* W2 = W0 + (size_t)32 * D_;
  const bf16_t* W3 = W0 + (size_t)48 * D_;

  v8f acc[2][4] = {};
  // buffer X holds step k, buffer Y holds step k+32
  v16bf xa0 = load_frag16(A0), xa1 = load_frag16(A1);
  v16bf xb0 = load_frag16(W0), xb1 = load_frag16(W1);
  v16bf xb2 = load_frag16(W2), xb3 = load_frag16(W3);
  v16bf ya0, ya1, yb0, yb1, yb2, yb3;

  for (int k0 = 0; k0 < D_ - 64; k0 += 64) {
    int k1 = k0 + 32, k2 = k0 + 64;
    __builtin_prefetch(A0 + k0 + 256, 0, 3);          // WGP-scope prefetch
    __builtin_prefetch(W0 + k0 + 256, 0, 3);
    ya0 = load_frag16(A0 + k1); ya1 = load_frag16(A1 + k1);
    yb0 = load_frag16(W0 + k1); yb1 = load_frag16(W1 + k1);
    yb2 = load_frag16(W2 + k1); yb3 = load_frag16(W3 + k1);
    acc[0][0] = wmma_bf16(xa0, xb0, acc[0][0]);
    acc[1][0] = wmma_bf16(xa1, xb0, acc[1][0]);
    acc[0][1] = wmma_bf16(xa0, xb1, acc[0][1]);
    acc[1][1] = wmma_bf16(xa1, xb1, acc[1][1]);
    acc[0][2] = wmma_bf16(xa0, xb2, acc[0][2]);
    acc[1][2] = wmma_bf16(xa1, xb2, acc[1][2]);
    acc[0][3] = wmma_bf16(xa0, xb3, acc[0][3]);
    acc[1][3] = wmma_bf16(xa1, xb3, acc[1][3]);
    xa0 = load_frag16(A0 + k2); xa1 = load_frag16(A1 + k2);
    xb0 = load_frag16(W0 + k2); xb1 = load_frag16(W1 + k2);
    xb2 = load_frag16(W2 + k2); xb3 = load_frag16(W3 + k2);
    acc[0][0] = wmma_bf16(ya0, yb0, acc[0][0]);
    acc[1][0] = wmma_bf16(ya1, yb0, acc[1][0]);
    acc[0][1] = wmma_bf16(ya0, yb1, acc[0][1]);
    acc[1][1] = wmma_bf16(ya1, yb1, acc[1][1]);
    acc[0][2] = wmma_bf16(ya0, yb2, acc[0][2]);
    acc[1][2] = wmma_bf16(ya1, yb2, acc[1][2]);
    acc[0][3] = wmma_bf16(ya0, yb3, acc[0][3]);
    acc[1][3] = wmma_bf16(ya1, yb3, acc[1][3]);
  }
  // tail: X holds k = D_-64 (preloaded in last iteration); Y takes D_-32.
  {
    int k1 = D_ - 32;
    ya0 = load_frag16(A0 + k1); ya1 = load_frag16(A1 + k1);
    yb0 = load_frag16(W0 + k1); yb1 = load_frag16(W1 + k1);
    yb2 = load_frag16(W2 + k1); yb3 = load_frag16(W3 + k1);
    acc[0][0] = wmma_bf16(xa0, xb0, acc[0][0]);
    acc[1][0] = wmma_bf16(xa1, xb0, acc[1][0]);
    acc[0][1] = wmma_bf16(xa0, xb1, acc[0][1]);
    acc[1][1] = wmma_bf16(xa1, xb1, acc[1][1]);
    acc[0][2] = wmma_bf16(xa0, xb2, acc[0][2]);
    acc[1][2] = wmma_bf16(xa1, xb2, acc[1][2]);
    acc[0][3] = wmma_bf16(xa0, xb3, acc[0][3]);
    acc[1][3] = wmma_bf16(xa1, xb3, acc[1][3]);
    acc[0][0] = wmma_bf16(ya0, yb0, acc[0][0]);
    acc[1][0] = wmma_bf16(ya1, yb0, acc[1][0]);
    acc[0][1] = wmma_bf16(ya0, yb1, acc[0][1]);
    acc[1][1] = wmma_bf16(ya1, yb1, acc[1][1]);
    acc[0][2] = wmma_bf16(ya0, yb2, acc[0][2]);
    acc[1][2] = wmma_bf16(ya1, yb2, acc[1][2]);
    acc[0][3] = wmma_bf16(ya0, yb3, acc[0][3]);
    acc[1][3] = wmma_bf16(ya1, yb3, acc[1][3]);
  }

#pragma unroll
  for (int mi = 0; mi < 2; ++mi) {
    int mb = m0 + wave * 32 + mi * 16 + g * 8;
#pragma unroll
    for (int s = 0; s < 4; ++s) {
      int n = n0 + s * 16 + r;
      float bv = bias[n];
#pragma unroll
      for (int v = 0; v < 8; ++v) {
        int m = mb + v;
        float val = acc[mi][s][v] + bv;
        if (mode == 0) {
          outB[((size_t)b * S_ + m) * D_ + n] = (bf16_t)val;
        } else if (mode == 1) {
          outB[((size_t)b * D_ + n) * S_ + m] = (bf16_t)val;
        } else {
          if (m < c) {
            int tok = idx[b * S_ + m];
            float w = gw[b * S_ + m];
            atomicAdd(&outScat[((size_t)b * S_ + tok) * D_ + n], w * val);
          }
        }
      }
    }
  }
}

// ---------------- flash attention over compacted tokens ---------------------
__global__ __launch_bounds__(128) void moe_attn(
    const bf16_t* __restrict__ Q, const bf16_t* __restrict__ K,
    const bf16_t* __restrict__ Vt, bf16_t* __restrict__ ctx,
    const int* __restrict__ cnt) {
  __shared__ bf16_t pbuf[4][16][64];
  int b = blockIdx.z, h = blockIdx.y;
  int c = cnt[b];
  int q0 = blockIdx.x * 64;
  if (q0 >= c) return;
  int wave = threadIdx.x >> 5, lane = threadIdx.x & 31;
  int r = lane & 15, g = lane >> 4;
  const bf16_t* Qp = Q + ((size_t)b * S_ + q0 + wave * 16 + r) * D_ + h * HD_ + g * 8;
  v16bf qf0 = load_frag16(Qp);
  v16bf qf1 = load_frag16(Qp + 32);
  const bf16_t* Kb = K + (size_t)b * S_ * D_ + h * HD_ + g * 8;
  const bf16_t* Vb = Vt + ((size_t)b * D_ + h * HD_) * S_ + g * 8;  // row d, col key
  v8f o[4] = {};
  float mrow[8], lrow[8];
#pragma unroll
  for (int v = 0; v < 8; ++v) { mrow[v] = -3.0e38f; lrow[v] = 0.f; }
  int ktiles = (c + 63) >> 6;
  for (int kt = 0; kt < ktiles; ++kt) {
    int kb = kt * 64;
    v8f sc[4] = {};
#pragma unroll
    for (int s = 0; s < 4; ++s) {
      const bf16_t* kp = Kb + (size_t)(kb + s * 16 + r) * D_;
      sc[s] = wmma_bf16(qf0, load_frag16(kp), sc[s]);
      sc[s] = wmma_bf16(qf1, load_frag16(kp + 32), sc[s]);
    }
    float tmax[8];
#pragma unroll
    for (int v = 0; v < 8; ++v) tmax[v] = -3.0e38f;
#pragma unroll
    for (int s = 0; s < 4; ++s) {
      bool valid = (kb + s * 16 + r) < c;
#pragma unroll
      for (int v = 0; v < 8; ++v) {
        float xs = valid ? sc[s][v] * 0.5f : -1.0e9f;   // sm_scale=0.5 + mask
        sc[s][v] = xs;
        tmax[v] = fmaxf(tmax[v], xs);
      }
    }
#pragma unroll
    for (int mk = 1; mk < 16; mk <<= 1)
#pragma unroll
      for (int v = 0; v < 8; ++v)
        tmax[v] = fmaxf(tmax[v], __shfl_xor(tmax[v], mk, 32));
    float fac[8];
#pragma unroll
    for (int v = 0; v < 8; ++v) {
      float mn = fmaxf(mrow[v], tmax[v]);
      fac[v] = __expf(mrow[v] - mn);
      mrow[v] = mn;
    }
    float rsum[8] = {0.f, 0.f, 0.f, 0.f, 0.f, 0.f, 0.f, 0.f};
#pragma unroll
    for (int s = 0; s < 4; ++s)
#pragma unroll
      for (int v = 0; v < 8; ++v) {
        float p = __expf(sc[s][v] - mrow[v]);
        sc[s][v] = p;
        rsum[v] += p;
      }
#pragma unroll
    for (int mk = 1; mk < 16; mk <<= 1)
#pragma unroll
      for (int v = 0; v < 8; ++v) rsum[v] += __shfl_xor(rsum[v], mk, 32);
#pragma unroll
    for (int v = 0; v < 8; ++v) lrow[v] = lrow[v] * fac[v] + rsum[v];
#pragma unroll
    for (int s = 0; s < 4; ++s)
#pragma unroll
      for (int v = 0; v < 8; ++v) o[s][v] *= fac[v];
    __syncthreads();
#pragma unroll
    for (int s = 0; s < 4; ++s)
#pragma unroll
      for (int v = 0; v < 8; ++v)
        pbuf[wave][v + g * 8][s * 16 + r] = (bf16_t)sc[s][v];
    __syncthreads();
    const bf16_t* pp = &pbuf[wave][r][g * 8];
    v16bf pf0 = load_frag16(pp);
    v16bf pf1 = load_frag16(pp + 32);
#pragma unroll
    for (int s = 0; s < 4; ++s) {
      const bf16_t* vp = Vb + (size_t)(s * 16 + r) * S_ + kb;
      o[s] = wmma_bf16(pf0, load_frag16(vp), o[s]);
      o[s] = wmma_bf16(pf1, load_frag16(vp + 32), o[s]);
    }
  }
  int mb = q0 + wave * 16 + g * 8;
#pragma unroll
  for (int s = 0; s < 4; ++s) {
    int d = s * 16 + r;
#pragma unroll
    for (int v = 0; v < 8; ++v) {
      float val = o[s][v] / lrow[v];
      ctx[((size_t)b * S_ + mb + v) * D_ + h * HD_ + d] = (bf16_t)val;
    }
  }
}

extern "C" void kernel_launch(void* const* d_in, const int* in_sizes, int n_in,
                              void* d_out, int out_size, void* d_ws, size_t ws_size,
                              hipStream_t stream) {
  (void)in_sizes; (void)n_in; (void)out_size; (void)ws_size;
  const float* x  = (const float*)d_in[0];
  const float* rw = (const float*)d_in[1];
  const float* rb = (const float*)d_in[2];
  const float* wq = (const float*)d_in[3];
  const float* bq = (const float*)d_in[4];
  const float* wk = (const float*)d_in[5];
  const float* bk = (const float*)d_in[6];
  const float* wv = (const float*)d_in[7];
  const float* bv = (const float*)d_in[8];
  const float* wo = (const float*)d_in[9];
  const float* bo = (const float*)d_in[10];
  float* out = (float*)d_out;

  char* ws = (char*)d_ws;
  size_t off = 0;
  auto alloc = [&](size_t bytes) -> void* {
    off = (off + 255) & ~(size_t)255;
    void* p = ws + off;
    off += bytes;
    return p;
  };
  float*  wtok = (float*)alloc((size_t)B_ * E_ * S_ * sizeof(float));
  int*    idx  = (int*)  alloc((size_t)B_ * S_ * sizeof(int));
  float*  gw   = (float*)alloc((size_t)B_ * S_ * sizeof(float));
  int*    cnt  = (int*)  alloc(256);
  bf16_t* WTq  = (bf16_t*)alloc((size_t)E_ * D_ * D_ * 2);
  bf16_t* WTk  = (bf16_t*)alloc((size_t)E_ * D_ * D_ * 2);
  bf16_t* WTv  = (bf16_t*)alloc((size_t)E_ * D_ * D_ * 2);
  bf16_t* WTo  = (bf16_t*)alloc((size_t)E_ * D_ * D_ * 2);
  bf16_t* xg   = (bf16_t*)alloc((size_t)B_ * S_ * D_ * 2);
  bf16_t* Qb   = (bf16_t*)alloc((size_t)B_ * S_ * D_ * 2);
  bf16_t* Kb   = (bf16_t*)alloc((size_t)B_ * S_ * D_ * 2);
  bf16_t* Vtb  = (bf16_t*)alloc((size_t)B_ * D_ * S_ * 2);
  bf16_t* ctx  = (bf16_t*)alloc((size_t)B_ * S_ * D_ * 2);

  hipMemsetAsync(d_out, 0, (size_t)B_ * S_ * D_ * sizeof(float), stream);

  moe_router<<<(B_ * S_) / 8, 256, 0, stream>>>(x, rw, rb, wtok);
  moe_wconvert<<<dim3(D_, E_), 128, 0, stream>>>(wq, WTq);
  moe_wconvert<<<dim3(D_, E_), 128, 0, stream>>>(wk, WTk);
  moe_wconvert<<<dim3(D_, E_), 128, 0, stream>>>(wv, WTv);
  moe_wconvert<<<dim3(D_, E_), 128, 0, stream>>>(wo, WTo);

  for (int e = 0; e < E_; ++e) {
    size_t wofs = (size_t)e * D_ * D_;
    moe_compact<<<B_, 32, 0, stream>>>(wtok, e, cnt, idx, gw);
    moe_gather<<<dim3(S_, B_), 128, 0, stream>>>(x, cnt, idx, xg);
    moe_gemm<<<dim3(16, 16, B_), 128, 0, stream>>>(
        xg, WTq + wofs, bq + e * D_, cnt, Qb, nullptr, nullptr, nullptr, 0);
    moe_gemm<<<dim3(16, 16, B_), 128, 0, stream>>>(
        xg, WTk + wofs, bk + e * D_, cnt, Kb, nullptr, nullptr, nullptr, 0);
    moe_gemm<<<dim3(16, 16, B_), 128, 0, stream>>>(
        xg, WTv + wofs, bv + e * D_, cnt, Vtb, nullptr, nullptr, nullptr, 1);
    moe_attn<<<dim3(32, 16, B_), 128, 0, stream>>>(Qb, Kb, Vtb, ctx, cnt);
    moe_gemm<<<dim3(16, 16, B_), 128, 0, stream>>>(
        ctx, WTo + wofs, bo + e * D_, cnt, nullptr, out, idx, gw, 2);
  }
}
